// AttentionLayer_9457517986403
// MI455X (gfx1250) — matrix operand order repaired and checked by reading, not compiled, not run on hardware
//
#include <hip/hip_runtime.h>
#include <hip/hip_bf16.h>

// ---------------------------------------------------------------------------
// AttentionLayer for MI455X (gfx1250, wave32, WMMA).
//   B=4, N=2048, C=256, H=8, HD=32.
//   Compute-bound (~21.5 GFLOP vs ~40MB traffic @ 23.3 TB/s) -> bf16 WMMA
//   (v_wmma_f32_16x16x32_bf16) with fp32 accumulation; fp32 softmax stats.
//   All fp32->bf16 conversions hoisted out of the GEMM hot loops.
// ---------------------------------------------------------------------------

typedef __attribute__((ext_vector_type(16))) __bf16 v16bf;
typedef __attribute__((ext_vector_type(4)))  __bf16 v4bf;
typedef __attribute__((ext_vector_type(8)))  float  v8f;

#define BB 4
#define NN 2048
#define CC 256
#define HH 8
#define HD 32
#define MM (BB * NN)   // 8192 rows

static __device__ __forceinline__ v8f wmma_bf16(v16bf a, v16bf b, v8f c) {
  // (neg_a, A, neg_b, B, c_mod, C, reuse_a, reuse_b)
  return __builtin_amdgcn_wmma_f32_16x16x32_bf16(false, a, false, b, (short)0, c,
                                                 false, false);
}

// ---- A operand (16x32, MxK, bf16). ISA 7.12.2: lanes 0-15 hold row M=lane,
// K in {0..7, 16..23}; lanes 16-31 hold row M=lane-16, K in {8..15, 24..31}.
// Two contiguous b128 loads per lane.
static __device__ __forceinline__ v16bf load_a_bf16(const __bf16* p, int ld, int lane) {
  int r  = lane & 15;
  int kb = (lane & 16) ? 8 : 0;
  const __bf16* base = p + (size_t)r * ld;
  v16bf a;
#pragma unroll
  for (int i = 0; i < 8; ++i) a[i] = base[kb + i];
#pragma unroll
  for (int i = 0; i < 8; ++i) a[8 + i] = base[16 + kb + i];
  return a;
}

// ---- B operand (32x16, KxN) from "transposed" storage w[n, k] (row-major,
// leading dim ld): B[k,n] = w[n,k]. Lanes 0-15: column n=lane, K in {0..7, 16..23};
// lanes 16-31: column n=lane-16, K in {8..15, 24..31}. Two b128 loads per lane.
static __device__ __forceinline__ v16bf load_bT_bf16(const __bf16* w, int ld, int lane) {
  int n  = lane & 15;
  int kb = (lane & 16) ? 8 : 0;
  const __bf16* base = w + (size_t)n * ld + kb;
  v16bf b;
#pragma unroll
  for (int i = 0; i < 8; ++i) b[i] = base[i];
#pragma unroll
  for (int i = 0; i < 8; ++i) b[8 + i] = base[16 + i];
  return b;
}

// Row reductions within a 16-lane half-wave (C/D rows are striped across
// lanes 0-15 / 16-31; xor masks < 16 keep the reduction inside each half).
static __device__ __forceinline__ float rowmax16(float v) {
#pragma unroll
  for (int m = 1; m < 16; m <<= 1) v = fmaxf(v, __shfl_xor(v, m, 32));
  return v;
}
static __device__ __forceinline__ float rowsum16(float v) {
#pragma unroll
  for (int m = 1; m < 16; m <<= 1) v += __shfl_xor(v, m, 32);
  return v;
}

// ===========================================================================
// Kernel 0: vectorized fp32 -> bf16 (one float4 -> 4x bf16 per thread).
// Grid sized exactly: element count must be a multiple of 1024.
// ===========================================================================
__global__ void __launch_bounds__(256) cvt_kernel(const float* __restrict__ s,
                                                  __bf16* __restrict__ d) {
  int i = (blockIdx.x * 256 + threadIdx.x) * 4;
  float4 f = *(const float4*)(s + i);
  v4bf o;
  o[0] = (__bf16)f.x; o[1] = (__bf16)f.y; o[2] = (__bf16)f.z; o[3] = (__bf16)f.w;
  *(v4bf*)(d + i) = o;
}

// ===========================================================================
// Kernel 1: qkv = x @ qkv_w.T (all-bf16 operands), scattered into bf16 head
// layout [B,H,N,HD]. One 16x64 strip per wave (4 accumulators).
// ===========================================================================
__global__ void __launch_bounds__(256) qkv_kernel(const __bf16* __restrict__ xb,
                                                  const __bf16* __restrict__ wb,
                                                  __bf16* __restrict__ qb,
                                                  __bf16* __restrict__ kb,
                                                  __bf16* __restrict__ vb) {
  const int lane = threadIdx.x & 31;
  const int wave = threadIdx.x >> 5;
  const int tile = blockIdx.x * 8 + wave;      // 6144 strips (512 x 12)
  const int mt = tile / 12, ng = tile % 12;
  const int m0 = mt * 16, n0 = ng * 64;

  v8f acc[4] = {};
#pragma unroll
  for (int kc = 0; kc < CC; kc += 32) {
    v16bf a = load_a_bf16(xb + (size_t)m0 * CC + kc, CC, lane);
#pragma unroll
    for (int t = 0; t < 4; ++t) {
      v16bf b = load_bT_bf16(wb + (size_t)(n0 + 16 * t) * CC + kc, CC, lane);
      acc[t] = wmma_bf16(a, b, acc[t]);
    }
  }

  // Scatter: global column -> {q|k|v}[b, h, n, hd] (bf16).
  const int hi = (lane >> 4) & 1;
#pragma unroll
  for (int t = 0; t < 4; ++t) {
    const int col   = n0 + 16 * t + (lane & 15);
    const int which = col >> 8;                // 0=q 1=k 2=v
    const int rem   = col & 255;
    const int h  = rem >> 5;
    const int hd = rem & 31;
    __bf16* dst = (which == 0) ? qb : (which == 1) ? kb : vb;
#pragma unroll
    for (int j = 0; j < 8; ++j) {
      int m    = m0 + j + 8 * hi;
      int bidx = m >> 11;                      // / N
      int nrow = m & (NN - 1);
      dst[((((size_t)bidx * HH + h) * NN) + nrow) * HD + hd] = (__bf16)acc[t][j];
    }
  }
}

// ===========================================================================
// Kernel 1b: transpose v [B,H,N,HD] -> vT [B,H,HD,N] via padded LDS tile so
// the attention O-GEMM B operand becomes contiguous b128 loads.
// One 64(n) x 32(hd) tile per 256-thread block; conflict-free both ways.
// ===========================================================================
__global__ void __launch_bounds__(256) vtrans_kernel(const __bf16* __restrict__ v,
                                                     __bf16* __restrict__ vT) {
  __shared__ __bf16 t[64 * 33];                // pad 33 to dodge bank conflicts
  const int bh = blockIdx.x >> 5;              // N/64 = 32 tiles per (b,h)
  const int n0 = (blockIdx.x & 31) * 64;
  const __bf16* src = v  + (size_t)bh * NN * HD + (size_t)n0 * HD;
  __bf16*       dst = vT + (size_t)bh * HD * NN + n0;
#pragma unroll
  for (int i = 0; i < 8; ++i) {                // coalesced reads along hd
    int idx = i * 256 + threadIdx.x;
    int nl = idx >> 5, hd = idx & 31;
    t[nl * 33 + hd] = src[(size_t)nl * HD + hd];
  }
  __syncthreads();
#pragma unroll
  for (int i = 0; i < 8; ++i) {                // coalesced writes along n
    int idx = i * 256 + threadIdx.x;
    int hd = idx >> 6, nl = idx & 63;
    dst[(size_t)hd * NN + nl] = t[nl * 33 + hd];
  }
}

// ===========================================================================
// Kernel 2: flash attention per (b, h, 16-row tile), 64-key chunks.
//   S(16x64) = 4 WMMA (K = HD = 32); O += P@V = 4 WMMA (two K=32 A-tiles).
// One wave per block -> LDS P-staging barriers are single-wave no-ops.
// Epilogue fuses the 0.1 * out + x residual and emits the bf16 A-operand for
// the output projection directly.
// ===========================================================================
__global__ void __launch_bounds__(32) attn_kernel(const __bf16* __restrict__ q,
                                                  const __bf16* __restrict__ k,
                                                  const __bf16* __restrict__ vT,
                                                  const float* __restrict__ adj,
                                                  const float* __restrict__ x,
                                                  __bf16* __restrict__ abf) {
  __shared__ __bf16 lds_p[16 * 64];            // P tile staging (C/D -> A relayout)

  const int lane = threadIdx.x;
  const int tile = blockIdx.x;                 // 4096 tiles
  const int bh = tile >> 7;                    // / (N/16)
  const int rt = tile & 127;
  const int b = bh >> 3, h = bh & 7;
  const int hi = (lane >> 4) & 1;
  const int ln = lane & 15;

  const __bf16* qp  = q  + (size_t)bh * NN * HD + (size_t)rt * 16 * HD;
  const __bf16* kp  = k  + (size_t)bh * NN * HD;
  const __bf16* vtp = vT + (size_t)bh * HD * NN;

  const v16bf aq = load_a_bf16(qp, HD, lane);  // 16 rows x 32 (full head dim)

  float m_i[8], l_i[8], alpha[8];
  v8f o0 = {}, o1 = {};                        // O cols 0-15 / 16-31
#pragma unroll
  for (int j = 0; j < 8; ++j) { m_i[j] = -3.0e38f; l_i[j] = 0.0f; }

  const float scale = 0.17677669529663689f;    // 1/sqrt(32)
  const float LOG2E = 1.4426950408889634f;

  for (int c = 0; c < NN; c += 64) {
    if (c + 64 < NN) {                         // uniform branch: EXEC unchanged
      __builtin_prefetch(kp + (size_t)(c + 64) * HD, 0, 0);   // global_prefetch
      __builtin_prefetch(vtp + (size_t)c + 64, 0, 0);
    }
    // S = q @ k^T for 64 keys (4 x 16-key B tiles).
    v8f z = {};
    v8f s[4];
#pragma unroll
    for (int t = 0; t < 4; ++t) {
      v16bf bk = load_bT_bf16(kp + (size_t)(c + 16 * t) * HD, HD, lane);
      s[t] = wmma_bf16(aq, bk, z);
    }

    // Online softmax with additive mask adj[row, key].
    const float* adjr = adj + (size_t)(rt * 16 + hi * 8) * NN + c + ln;
#pragma unroll
    for (int j = 0; j < 8; ++j) {
      float a0 = s[0][j] * scale + adjr[(size_t)j * NN];
      float a1 = s[1][j] * scale + adjr[(size_t)j * NN + 16];
      float a2 = s[2][j] * scale + adjr[(size_t)j * NN + 32];
      float a3 = s[3][j] * scale + adjr[(size_t)j * NN + 48];
      float mx = rowmax16(fmaxf(fmaxf(a0, a1), fmaxf(a2, a3)));
      float mnew = fmaxf(m_i[j], mx);
      float e0 = __builtin_exp2f((a0 - mnew) * LOG2E);
      float e1 = __builtin_exp2f((a1 - mnew) * LOG2E);
      float e2 = __builtin_exp2f((a2 - mnew) * LOG2E);
      float e3 = __builtin_exp2f((a3 - mnew) * LOG2E);
      float al = __builtin_exp2f((m_i[j] - mnew) * LOG2E);
      l_i[j] = l_i[j] * al + rowsum16((e0 + e1) + (e2 + e3));
      m_i[j] = mnew;
      alpha[j] = al;
      int row = j + 8 * hi;                    // stage P (C/D -> LDS row-major)
      lds_p[row * 64 + ln]      = (__bf16)e0;
      lds_p[row * 64 + 16 + ln] = (__bf16)e1;
      lds_p[row * 64 + 32 + ln] = (__bf16)e2;
      lds_p[row * 64 + 48 + ln] = (__bf16)e3;
    }
    __syncthreads();                            // single-wave: waitcnt only
    v16bf ap0 = load_a_bf16(lds_p, 64, lane);       // P keys c..c+31
    v16bf ap1 = load_a_bf16(lds_p + 32, 64, lane);  // P keys c+32..c+63

    // Rescale O by alpha, then O += P @ V (vT gives contiguous B loads).
#pragma unroll
    for (int j = 0; j < 8; ++j) { o0[j] *= alpha[j]; o1[j] *= alpha[j]; }
    v16bf bv00 = load_bT_bf16(vtp + c,                      NN, lane); // hd0-15,k0-31
    v16bf bv10 = load_bT_bf16(vtp + c + 32,                 NN, lane); // hd0-15,k32-63
    v16bf bv01 = load_bT_bf16(vtp + (size_t)16 * NN + c,      NN, lane);
    v16bf bv11 = load_bT_bf16(vtp + (size_t)16 * NN + c + 32, NN, lane);
    o0 = wmma_bf16(ap0, bv00, o0);
    o0 = wmma_bf16(ap1, bv10, o0);
    o1 = wmma_bf16(ap0, bv01, o1);
    o1 = wmma_bf16(ap1, bv11, o1);
    __syncthreads();
  }

  // Epilogue: normalize, fuse 0.1*out + x, store bf16 A-operand [B, N, C].
  const size_t rowbase = ((size_t)b * NN + rt * 16) * CC + h * HD;
  const float* xr = x   + rowbase;
  __bf16*      op = abf + rowbase;
#pragma unroll
  for (int j = 0; j < 8; ++j) {
    int row = j + 8 * hi;
    float inv = 0.1f / l_i[j];
    op[(size_t)row * CC + ln]      = (__bf16)(o0[j] * inv + xr[(size_t)row * CC + ln]);
    op[(size_t)row * CC + 16 + ln] = (__bf16)(o1[j] * inv + xr[(size_t)row * CC + 16 + ln]);
  }
}

// ===========================================================================
// Kernel 3: y = abf @ out_w.T + out_b   (bf16 operands, fp32 output)
// One 16x64 strip per wave.
// ===========================================================================
__global__ void __launch_bounds__(256) out_proj_kernel(const __bf16* __restrict__ abf,
                                                       const __bf16* __restrict__ owb,
                                                       const float* __restrict__ out_b,
                                                       float* __restrict__ y) {
  const int lane = threadIdx.x & 31;
  const int wave = threadIdx.x >> 5;
  const int tile = blockIdx.x * 8 + wave;      // 2048 strips (512 x 4)
  const int mt = tile >> 2, ng = tile & 3;
  const int m0 = mt * 16, n0 = ng * 64;

  v8f acc[4] = {};
#pragma unroll
  for (int kc = 0; kc < CC; kc += 32) {
    v16bf a = load_a_bf16(abf + (size_t)m0 * CC + kc, CC, lane);
#pragma unroll
    for (int t = 0; t < 4; ++t) {
      v16bf b = load_bT_bf16(owb + (size_t)(n0 + 16 * t) * CC + kc, CC, lane);
      acc[t] = wmma_bf16(a, b, acc[t]);
    }
  }

  const int hi = (lane >> 4) & 1;
#pragma unroll
  for (int t = 0; t < 4; ++t) {
    const int col = n0 + 16 * t + (lane & 15);
    const float bias = out_b[col];
#pragma unroll
    for (int j = 0; j < 8; ++j) {
      int m = m0 + j + 8 * hi;
      y[(size_t)m * CC + col] = acc[t][j] + bias;
    }
  }
}

// ===========================================================================
extern "C" void kernel_launch(void* const* d_in, const int* in_sizes, int n_in,
                              void* d_out, int out_size, void* d_ws, size_t ws_size,
                              hipStream_t stream) {
  // setup_inputs order: x, x0, adj, qkv_w, out_w, out_b, pos_w, pos_b
  const float* x     = (const float*)d_in[0];
  const float* adj   = (const float*)d_in[2];
  const float* qkv_w = (const float*)d_in[3];
  const float* out_w = (const float*)d_in[4];
  const float* out_b = (const float*)d_in[5];
  float* y = (float*)d_out;

  // Workspace carve-up (all bf16, ~24.5 MB total).
  const size_t BHND = (size_t)BB * HH * NN * HD;   // 2,097,152 elems
  __bf16* qb    = (__bf16*)d_ws;                   // 4 MB
  __bf16* kb    = qb + BHND;                       // 4 MB
  __bf16* vb    = kb + BHND;                       // 4 MB  [B,H,N,HD]
  __bf16* vTb   = vb + BHND;                       // 4 MB  [B,H,HD,N]
  __bf16* owb   = vTb + BHND;                      // 128 KB
  __bf16* qkvwb = owb + (size_t)CC * CC;           // 384 KB
  __bf16* xb    = qkvwb + (size_t)3 * CC * CC;     // 4 MB  [B*N, C]
  __bf16* abf   = xb + (size_t)MM * CC;            // 4 MB  [B*N, C] fused A

  // 0) One-time fp32 -> bf16 conversions (counts all multiples of 1024).
  cvt_kernel<<<(CC * CC) / 1024, 256, 0, stream>>>(out_w, owb);
  cvt_kernel<<<(3 * CC * CC) / 1024, 256, 0, stream>>>(qkv_w, qkvwb);
  cvt_kernel<<<(MM * CC) / 1024, 256, 0, stream>>>(x, xb);

  // 1) QKV projection into bf16 head layouts (pure-bf16 GEMM).
  qkv_kernel<<<768, 256, 0, stream>>>(xb, qkvwb, qb, kb, vb);
  // 1b) v -> vT for contiguous WMMA B-operand loads.
  vtrans_kernel<<<BB * HH * (NN / 64), 256, 0, stream>>>(vb, vTb);
  // 2) flash attention; epilogue fuses 0.1*out + x into bf16 A-operand.
  attn_kernel<<<BB * HH * (NN / 16), 32, 0, stream>>>(qb, kb, vTb, adj, x, abf);
  // 3) output projection (pure-bf16 GEMM, fp32 epilogue with bias).
  out_proj_kernel<<<256, 256, 0, stream>>>(abf, owb, out_b, y);
}
